// BinaryTreeLSTM_82240033784212
// MI455X (gfx1250) — matrix operand order repaired and checked by reading, not compiled, not run on hardware
//
#include <hip/hip_runtime.h>
#include <hip/hip_bf16.h>
#include <cstdint>
#include <cstddef>

// Shapes
#define BB    64
#define LL    1024
#define DIMM  256
#define NCOL  1024          // 4 live gate blocks (i, f1, o, g) * 256
#define OUTL  2047          // L + (L-1)

typedef __bf16 v16bf __attribute__((ext_vector_type(16)));
typedef __bf16 v8bf  __attribute__((ext_vector_type(8)));
typedef float  v8f   __attribute__((ext_vector_type(8)));

// ---- workspace layout (bytes) ----
static const size_t OFF_BAR  = 0;                              // 1 u32, padded
static const size_t OFF_W1B  = 256;                            // 256x1024 bf16 (h-half of W, live gates)
static const size_t OFF_W2B  = OFF_W1B + 256 * 1024 * 2;       // 256x1024 bf16 (x-half of W)
static const size_t OFF_HBUF = OFF_W2B + 256 * 1024 * 2;       // 2 x 64x256 bf16 (ping-pong h)
static const size_t OFF_CBUF = OFF_HBUF + 2 * BB * DIMM * 2;   // 64x256 f32 (c state)
static const size_t OFF_P    = OFF_CBUF + BB * DIMM * 4;       // 65536x1024 f32: x@W2 + b  (256 MB)
// total ws needed ~ 269.6 MB

__device__ __forceinline__ float sigmoidf_(float x) { return 1.0f / (1.0f + __expf(-x)); }

// live-gate u (0..3 = i,f1,o,g) -> column block in original W (0,1,3,4; block 2 = dead f2)
__device__ __forceinline__ int gate_col_block(int u) { return u + (u >= 2 ? 1 : 0); }

// A-fragment K index for v_wmma_*_16x16x32_bf16 (ISA 7.12.2, 16-bit A 16x32)
__device__ __forceinline__ int a_k_of(int kk, int j, int half) {
    return kk * 32 + ((j >> 2) << 4) + (half << 3) + ((j & 3) << 1);
}

// ---------------- prep kernels ----------------

__global__ void lstm_pack_w(const float* __restrict__ W, __bf16* __restrict__ W1b,
                            __bf16* __restrict__ W2b) {
    int idx = blockIdx.x * blockDim.x + threadIdx.x;     // 0 .. 256*1024-1
    int k   = idx >> 10;
    int col = idx & 1023;
    int u   = col >> 8;
    int gcol = gate_col_block(u) * 256 + (col & 255);
    W1b[idx] = (__bf16)W[(size_t)k * 1280 + gcol];          // rows 0..255   multiply h
    W2b[idx] = (__bf16)W[(size_t)(256 + k) * 1280 + gcol];  // rows 256..511 multiply x_t
}

__global__ void lstm_init_state(const float* __restrict__ x, __bf16* __restrict__ hbuf,
                                float* __restrict__ cbuf, unsigned int* __restrict__ bar) {
    int idx = blockIdx.x * blockDim.x + threadIdx.x;     // 0 .. 64*256-1
    int b = idx >> 8, d = idx & 255;
    hbuf[idx] = (__bf16)x[(size_t)b * LL * DIMM + d];    // h0 = x[:,0]
    cbuf[idx] = 0.0f;
    if (idx == 0) *bar = 0u;
}

__global__ void lstm_copy_x(const float* __restrict__ x, float* __restrict__ out) {
    size_t idx = (size_t)blockIdx.x * blockDim.x + threadIdx.x;  // over 4M float4
    size_t b   = idx >> 16;                  // 1024*256/4 = 65536 float4 per batch
    size_t rem = idx & 65535;
    const float4* src = (const float4*)x;
    float4* dst = (float4*)out;
    dst[b * ((size_t)OUTL * DIMM / 4) + rem] = src[idx];
}

// ---------------- phase 1: P = x @ W2 + b over all (b,t) ----------------
// grid: 4096 WGs (one 16-row M-tile each), 256 threads = 8 waves, wave handles 8 N-tiles.

__global__ void lstm_precompute(const float* __restrict__ x, const __bf16* __restrict__ W2b,
                                const float* __restrict__ bias, float* __restrict__ P) {
    __shared__ __bf16 lds[16 * 256];
    const int tid = threadIdx.x;
    const int m0  = blockIdx.x * 16;

    {   // 16x256 fp32 tile of x -> bf16 LDS
        int row = tid >> 4;
        int ch  = (tid & 15) * 16;
        const float* src = x + (size_t)(m0 + row) * 256 + ch;
        #pragma unroll
        for (int i = 0; i < 16; i += 4) {
            float4 v = *(const float4*)(src + i);
            lds[row * 256 + ch + i + 0] = (__bf16)v.x;
            lds[row * 256 + ch + i + 1] = (__bf16)v.y;
            lds[row * 256 + ch + i + 2] = (__bf16)v.z;
            lds[row * 256 + ch + i + 3] = (__bf16)v.w;
        }
    }
    __syncthreads();

    const int lane = tid & 31, wave = tid >> 5;
    const int m = lane & 15, half = lane >> 4, n = lane & 15;

    v16bf a[8];                                   // A-fragments for the 8 K-steps
    #pragma unroll
    for (int kk = 0; kk < 8; ++kk)
        #pragma unroll
        for (int j = 0; j < 8; ++j) {
            int K = a_k_of(kk, j, half);
            a[kk][2 * j]     = lds[m * 256 + K];
            a[kk][2 * j + 1] = lds[m * 256 + K + 1];
        }

    #pragma unroll 1
    for (int q = 0; q < 8; ++q) {
        int nt = wave * 8 + q;                    // 0..63
        v8f acc = {};
        #pragma unroll
        for (int kk = 0; kk < 8; ++kk) {
            v16bf bf;
            #pragma unroll
            for (int s = 0; s < 16; ++s)
                bf[s] = W2b[(size_t)(kk * 32 + ((s >> 3) << 4) + 8 * half + (s & 7)) * NCOL + nt * 16 + n];
            acc = __builtin_amdgcn_wmma_f32_16x16x32_bf16(false, a[kk], false, bf,
                                                          (short)0, acc, false, false);
        }
        int col = nt * 16 + n;
        int u = col >> 8;
        float bb = bias[gate_col_block(u) * 256 + (col & 255)];
        #pragma unroll
        for (int r = 0; r < 8; ++r)
            P[(size_t)(m0 + r + 8 * half) * NCOL + col] = acc[r] + bb;
    }
}

// ---------------- phase 2: persistent recurrent scan ----------------
// 16 WGs x 128 thr = 64 waves; wave (mt,dt) owns the 16x16 (batch x dim) state tile.
// All W1b B-fragments (4 gates x 8 k-steps x 8 VGPRs = 256 VGPRs) are hoisted into
// registers for the whole scan (CDNA5 wave32 addresses up to 1024 VGPRs via VGPR-MSB).

__global__ void lstm_scan(const __bf16* __restrict__ W1b, const float* __restrict__ P,
                          __bf16* __restrict__ hbuf, float* __restrict__ cbuf,
                          float* __restrict__ out, unsigned int* __restrict__ bar) {
    const int tid  = threadIdx.x;
    const int lane = tid & 31;
    const int wgid = blockIdx.x * 4 + (tid >> 5);   // 0..63
    const int mt = wgid >> 4, dt = wgid & 15;
    const int m = lane & 15, half = lane >> 4, n = lane & 15;
    const int arow = mt * 16 + m;                   // batch row this lane gathers for A
    const int dcol = dt * 16 + n;

    // loop-invariant B operands: W1b tile columns for this wave's (gate u, dt)
    v16bf Bw[4][8];
    #pragma unroll
    for (int u = 0; u < 4; ++u)
        #pragma unroll
        for (int kk = 0; kk < 8; ++kk)
            for (int s = 0; s < 16; ++s)
                Bw[u][kk][s] = W1b[(size_t)(kk * 32 + ((s >> 3) << 4) + 8 * half + (s & 7)) * NCOL
                                   + u * 256 + dt * 16 + n];

    for (int t = 1; t < LL; ++t) {
        const __bf16* hsrc = hbuf + ((t - 1) & 1) * (BB * DIMM);
        __bf16*       hdst = hbuf + (t & 1)       * (BB * DIMM);

        // A-fragments: two contiguous 16B chunks per k-step (ISA 16-bit A layout)
        v16bf a[8];
        #pragma unroll
        for (int kk = 0; kk < 8; ++kk) {
            v8bf lo = *(const v8bf*)(hsrc + arow * 256 + kk * 32 + 8 * half);
            v8bf hi = *(const v8bf*)(hsrc + arow * 256 + kk * 32 + 16 + 8 * half);
            #pragma unroll
            for (int e = 0; e < 8; ++e) { a[kk][e] = lo[e]; a[kk][8 + e] = hi[e]; }
        }

        v8f acc[4];
        #pragma unroll
        for (int u = 0; u < 4; ++u) {
            v8f z = {};
            acc[u] = z;
            #pragma unroll
            for (int kk = 0; kk < 8; ++kk)
                acc[u] = __builtin_amdgcn_wmma_f32_16x16x32_bf16(false, a[kk], false, Bw[u][kk],
                                                                 (short)0, acc[u], false, false);
        }

        #pragma unroll
        for (int r = 0; r < 8; ++r) {
            int b = mt * 16 + r + 8 * half;
            size_t prow = ((size_t)b * LL + t) * NCOL;
            float si = acc[0][r] + P[prow +       dcol];
            float sf = acc[1][r] + P[prow + 256 + dcol];
            float so = acc[2][r] + P[prow + 512 + dcol];
            float sg = acc[3][r] + P[prow + 768 + dcol];
            float c_new = sigmoidf_(sf) * cbuf[b * 256 + dcol] + sigmoidf_(si) * tanhf(sg);
            float h     = sigmoidf_(so) * tanhf(c_new);
            cbuf[b * 256 + dcol] = c_new;
            out[((size_t)b * OUTL + (LL - 1) + t) * DIMM + dcol] = h;
            hdst[b * 256 + dcol] = (__bf16)h;
        }

        // hide next step's HBM latency: prefetch P slice for t+1 (global_prefetch_b8)
        if (t + 1 < LL) {
            #pragma unroll
            for (int r = 0; r < 8; ++r) {
                int b = mt * 16 + r + 8 * half;
                const float* pn = P + ((size_t)b * LL + (t + 1)) * NCOL + dcol;
                __builtin_prefetch(pn,       0, 1);
                __builtin_prefetch(pn + 256, 0, 1);
                __builtin_prefetch(pn + 512, 0, 1);
                __builtin_prefetch(pn + 768, 0, 1);
            }
        }

        // grid-wide barrier: release our stores, arrive, spin, acquire others' stores
        __builtin_amdgcn_fence(__ATOMIC_RELEASE, "agent");
        __syncthreads();
        if (tid == 0) {
            unsigned v = __hip_atomic_fetch_add(bar, 1u, __ATOMIC_ACQ_REL,
                                                __HIP_MEMORY_SCOPE_AGENT);
            unsigned target = (v / 16u + 1u) * 16u;
            while (__hip_atomic_load(bar, __ATOMIC_ACQUIRE, __HIP_MEMORY_SCOPE_AGENT) < target)
                __builtin_amdgcn_s_sleep(1);
        }
        __syncthreads();
        __builtin_amdgcn_fence(__ATOMIC_ACQUIRE, "agent");
    }
}

// ---------------- launch ----------------

extern "C" void kernel_launch(void* const* d_in, const int* in_sizes, int n_in,
                              void* d_out, int out_size, void* d_ws, size_t ws_size,
                              hipStream_t stream) {
    const float* x    = (const float*)d_in[0];   // (64,1024,256) f32
    const float* W    = (const float*)d_in[1];   // (512,1280) f32
    const float* bias = (const float*)d_in[2];   // (1280,) f32
    (void)in_sizes; (void)n_in; (void)out_size; (void)ws_size;

    char* ws = (char*)d_ws;
    unsigned int* bar  = (unsigned int*)(ws + OFF_BAR);
    __bf16*       W1b  = (__bf16*)(ws + OFF_W1B);
    __bf16*       W2b  = (__bf16*)(ws + OFF_W2B);
    __bf16*       hbuf = (__bf16*)(ws + OFF_HBUF);
    float*        cbuf = (float*)(ws + OFF_CBUF);
    float*        P    = (float*)(ws + OFF_P);
    float*        out  = (float*)d_out;

    lstm_pack_w     <<<1024,  256, 0, stream>>>(W, W1b, W2b);
    lstm_init_state <<<64,    256, 0, stream>>>(x, hbuf, cbuf, bar);
    lstm_copy_x     <<<16384, 256, 0, stream>>>(x, out);
    lstm_precompute <<<4096,  256, 0, stream>>>(x, W2b, bias, P);
    lstm_scan       <<<16,    128, 0, stream>>>(W1b, P, hbuf, cbuf, out, bar);
}
